// ms_gcn_group_cat_73177652789791
// MI455X (gfx1250) — compile-verified, matrix operand order checked by reading, not run on hardware
//
#include <hip/hip_runtime.h>
#include <hip/hip_fp16.h>

typedef __attribute__((ext_vector_type(16))) _Float16 v16h;
typedef __attribute__((ext_vector_type(8)))  _Float16 v8h;
typedef __attribute__((ext_vector_type(8)))  float    v8f;
typedef __attribute__((ext_vector_type(4)))  int      v4i;

#define S_    8
#define V_    25
#define CIN_  64
#define COUT_ 64
#define NB_   32
#define T_    128
#define C2_   128            // 2*CIN
#define KDIM  (C2_*V_)       // 3200
#define MDIM  (COUT_*V_)     // 1600
#define NTDIM (NB_*T_)       // 4096
#define BN_EPS 1e-5f

// ---- CDNA5 async direct-to-LDS staging (guarded; falls back to sync staging) ----
#if defined(__gfx1250__) && __has_builtin(__builtin_amdgcn_global_load_async_to_lds_b128)
#define USE_ASYNC_LDS 1
typedef __attribute__((address_space(1))) v4i g_v4i;   // global int4
typedef __attribute__((address_space(3))) v4i l_v4i;   // LDS int4
#define ASYNC_B128(g, l) \
    __builtin_amdgcn_global_load_async_to_lds_b128((g_v4i*)(g), (l_v4i*)(l), 0, 0)
#if __has_builtin(__builtin_amdgcn_s_wait_asynccnt)
#define WAIT_ASYNC(n) __builtin_amdgcn_s_wait_asynccnt(n)
#else
#define WAIT_ASYNC(n) asm volatile("s_wait_asynccnt %0" :: "i"(n) : "memory")
#endif
#else
#define USE_ASYNC_LDS 0
#endif

// ---------------- Kernel 1: temporal stencil + concat -> H (fp16, NT-major [nt][k]) --
__global__ void prep_h_kernel(const float* __restrict__ x, _Float16* __restrict__ H) {
    int idx = blockIdx.x * blockDim.x + threadIdx.x;
    const int total = KDIM * NTDIM;
    if (idx >= total) return;
    int nt = idx / KDIM;
    int k  = idx - nt * KDIM;
    int c  = k / V_;
    int u  = k - c * V_;
    int n  = nt >> 7;          // / T_
    int t  = nt & (T_ - 1);
    float val;
    if (c < CIN_) {
        val = x[((n * CIN_ + c) * T_ + t) * V_ + u];
    } else {
        int c0 = c - CIN_;
        const float* xb = x + (size_t)((n * CIN_ + c0) * T_) * V_ + u;
        // diff(t) = x[t]-x[t-1] with x[-1]->0 (diff[0]=x[0]); diff outside [0,T) is 0 (avg pad)
        auto xat = [&](int tt) -> float {
            return (tt >= 0 && tt < T_) ? xb[tt * V_] : 0.f;
        };
        auto dif = [&](int tt) -> float {
            return (tt >= 0 && tt < T_) ? (xat(tt) - xat(tt - 1)) : 0.f;
        };
        val = (dif(t - 1) + dif(t) + dif(t + 1)) * (1.f / 3.f);
    }
    H[idx] = (_Float16)val;
}

// ---------------- Kernel 2: combined weight WA[(o,v)][(c,u)] = sum_s W[o,s*128+c]*(A+B)[s*25+v,u]
__global__ void prep_wa_kernel(const float* __restrict__ W, const float* __restrict__ A,
                               const float* __restrict__ Bm, _Float16* __restrict__ WA) {
    int idx = blockIdx.x * blockDim.x + threadIdx.x;
    const int total = MDIM * KDIM;
    if (idx >= total) return;
    int r  = idx / KDIM;
    int kk = idx - r * KDIM;
    int o  = r / V_;
    int v  = r - o * V_;
    int c  = kk / V_;
    int u  = kk - c * V_;
    float acc = 0.f;
#pragma unroll
    for (int s = 0; s < S_; ++s) {
        float w  = W[o * (S_ * C2_) + s * C2_ + c];
        int row  = (s * V_ + v) * V_ + u;
        acc += w * (A[row] + Bm[row]);
    }
    WA[idx] = (_Float16)acc;
}

// ---------------- Kernel 3: Out(1600x4096) = WA(1600x3200) @ H^T + fused BN/ReLU
// block = 128 threads = 4 wave32; block tile 64(M) x 64(N); K step = 32; LDS double-buffered.
__global__ __launch_bounds__(128)
void fused_gemm_kernel(const _Float16* __restrict__ WA, const _Float16* __restrict__ H,
                       const float* __restrict__ bias,  const float* __restrict__ gamma,
                       const float* __restrict__ beta,  const float* __restrict__ mean,
                       const float* __restrict__ var,   float* __restrict__ out) {
    __shared__ __align__(16) _Float16 As[2][64][32];   // [buf][m][k]
    __shared__ __align__(16) _Float16 Bs[2][64][32];   // [buf][n][k]

    const int tid  = threadIdx.x;
    const int lane = tid & 31;
    const int wave = tid >> 5;
    const int m0   = blockIdx.y * 64;
    const int n0   = blockIdx.x * 64;

    v8f acc[4] = {};   // 4 M-tiles of 16x16 fp32 accumulators per wave

    // loader coordinates: each thread moves 32 bytes of A and 32 bytes of B per stage
    const int l_row = tid >> 1;          // 0..63 (m for A, n for B)
    const int l_k   = (tid & 1) * 16;    // 0 or 16

    const int kbase = (lane < 16) ? 0 : 8;      // WMMA 16-bit A/B per-lane K grouping
    const int nloc  = (wave << 4) + (lane & 15);

    auto stage = [&](int k0, int buf) {
        const _Float16* ga = WA + (size_t)(m0 + l_row) * KDIM + k0 + l_k;
        const _Float16* gb = H  + (size_t)(n0 + l_row) * KDIM + k0 + l_k;
#if USE_ASYNC_LDS
        ASYNC_B128(ga,     &As[buf][l_row][l_k]);
        ASYNC_B128(ga + 8, &As[buf][l_row][l_k + 8]);
        ASYNC_B128(gb,     &Bs[buf][l_row][l_k]);
        ASYNC_B128(gb + 8, &Bs[buf][l_row][l_k + 8]);
#else
        *(v8h*)&As[buf][l_row][l_k]     = *(const v8h*)(ga);
        *(v8h*)&As[buf][l_row][l_k + 8] = *(const v8h*)(ga + 8);
        *(v8h*)&Bs[buf][l_row][l_k]     = *(const v8h*)(gb);
        *(v8h*)&Bs[buf][l_row][l_k + 8] = *(const v8h*)(gb + 8);
#endif
        __builtin_prefetch(ga + 64, 0, 3);   // next-next K tile -> global_prefetch_b8
        __builtin_prefetch(gb + 64, 0, 3);
    };

    const int NK = KDIM / 32;     // 100
    stage(0, 0);
    for (int i = 0; i < NK; ++i) {
        const int buf = i & 1;
        if (i + 1 < NK) stage((i + 1) * 32, buf ^ 1);
#if USE_ASYNC_LDS
        // async copies complete in order: <=4 outstanding => this tile's 4 copies landed
        if (i + 1 < NK) { WAIT_ASYNC(4); } else { WAIT_ASYNC(0); }
#endif
        __syncthreads();   // all waves' tile-i data visible in LDS

        // B fragment: column nloc; halves[0..7]=K kbase..+7, halves[8..15]=K kbase+16..+23
        v16h bfrag;
        {
            v8h blo = *(const v8h*)&Bs[buf][nloc][kbase];
            v8h bhi = *(const v8h*)&Bs[buf][nloc][kbase + 16];
#pragma unroll
            for (int j = 0; j < 8; ++j) { bfrag[j] = blo[j]; bfrag[8 + j] = bhi[j]; }
        }
#pragma unroll
        for (int mt = 0; mt < 4; ++mt) {
            const int mloc = (mt << 4) + (lane & 15);
            v16h afrag;
            v8h alo = *(const v8h*)&As[buf][mloc][kbase];
            v8h ahi = *(const v8h*)&As[buf][mloc][kbase + 16];
#pragma unroll
            for (int j = 0; j < 8; ++j) { afrag[j] = alo[j]; afrag[8 + j] = ahi[j]; }
            acc[mt] = __builtin_amdgcn_wmma_f32_16x16x32_f16(
                false, afrag, false, bfrag, (short)0, acc[mt], false, false);
        }
        __syncthreads();   // all waves done reading buf before it is overwritten
    }

    // epilogue: D layout -> lane<16: (M = r, N = lane); lane>=16: (M = 8+r, N = lane-16)
    const int gc = n0 + (wave << 4) + (lane & 15);   // nt column
    const int nn = gc >> 7;
    const int tt = gc & 127;
    const int mofs = (lane < 16) ? 0 : 8;
#pragma unroll
    for (int mt = 0; mt < 4; ++mt) {
#pragma unroll
        for (int r = 0; r < 8; ++r) {
            int gr = m0 + (mt << 4) + mofs + r;      // row = o*25 + v
            int o  = gr / V_;
            int v  = gr - o * V_;
            float inv   = gamma[o] * rsqrtf(var[o] + BN_EPS);
            float shift = bias[o] * inv + beta[o] - mean[o] * inv;
            float val   = acc[mt][r] * inv + shift;
            val = val > 0.f ? val : 0.f;
            out[(size_t)((nn * COUT_ + o) * T_ + tt) * V_ + v] = val;
        }
    }
}

extern "C" void kernel_launch(void* const* d_in, const int* in_sizes, int n_in,
                              void* d_out, int out_size, void* d_ws, size_t ws_size,
                              hipStream_t stream) {
    const float* x      = (const float*)d_in[0];
    const float* A_norm = (const float*)d_in[1];
    const float* Bm     = (const float*)d_in[2];
    const float* W      = (const float*)d_in[3];
    const float* b      = (const float*)d_in[4];
    const float* gamma  = (const float*)d_in[5];
    const float* beta   = (const float*)d_in[6];
    const float* rmean  = (const float*)d_in[7];
    const float* rvar   = (const float*)d_in[8];
    float* out = (float*)d_out;

    _Float16* H  = (_Float16*)d_ws;                                      // 4096*3200*2 = 26.2 MB
    _Float16* WA = (_Float16*)((char*)d_ws + (size_t)KDIM * NTDIM * 2);  // 1600*3200*2 = 10.2 MB

    {
        int total = KDIM * NTDIM;
        prep_h_kernel<<<(total + 255) / 256, 256, 0, stream>>>(x, H);
    }
    {
        int total = MDIM * KDIM;
        prep_wa_kernel<<<(total + 255) / 256, 256, 0, stream>>>(W, A_norm, Bm, WA);
    }
    {
        dim3 grid(NTDIM / 64, MDIM / 64);   // (64, 25)
        fused_gemm_kernel<<<grid, 128, 0, stream>>>(WA, H, b, gamma, beta, rmean, rvar, out);
    }
}
